// Encoder_80736795230985
// MI455X (gfx1250) — matrix operand order repaired and checked by reading, not compiled, not run on hardware
//
#include <hip/hip_runtime.h>

// ---------------------------------------------------------------------------
// Spiking encoder on MI455X (gfx1250):
//   per step t: h = x_t @ W1^T ; LIF -> s1 ; o = s1 @ W2^T ; LIF -> s2
// bf16 WMMA (spikes are binary -> exact in bf16), f32 accumulate,
// double-buffered LDS staged with GLOBAL_LOAD_ASYNC_TO_LDS (ASYNCcnt).
// ---------------------------------------------------------------------------

typedef __attribute__((ext_vector_type(16))) __bf16 v16bf;
typedef __attribute__((ext_vector_type(8)))  __bf16 v8bf;
typedef __attribute__((ext_vector_type(4)))  __bf16 v4bf;
typedef __attribute__((ext_vector_type(8)))  float  v8f;
typedef __attribute__((ext_vector_type(4)))  float  v4f;
typedef __attribute__((ext_vector_type(4)))  int    v4i;

// Pointer types matching the async builtin's signature (from hipcc diagnostic:
// param 0 is 'int __vector(4) __device__ *', i.e. v4i in addrspace(1)).
typedef __attribute__((address_space(1))) v4i* as1_v4i_ptr;
typedef __attribute__((address_space(3))) v4i* as3_v4i_ptr;

#define B_DIM  512
#define T_DIM  16
#define FIN    2048
#define HID    2048
#define OUTN   1024

constexpr int BKT   = 64;        // K tile per stage
constexpr int PITCH = BKT + 8;   // 144B row pitch: pads LDS banks, keeps 16B align

// 16B global -> LDS copy; async path if the toolchain exposes the builtin.
__device__ __forceinline__ void cp16(const __bf16* g, __bf16* l) {
#if __has_builtin(__builtin_amdgcn_global_load_async_to_lds_b128)
    __builtin_amdgcn_global_load_async_to_lds_b128(
        (as1_v4i_ptr)g, (as3_v4i_ptr)l, 0, 0);
#else
    *(v8bf*)l = *(const v8bf*)g;      // sync fallback through VGPRs
#endif
}

template <int N>
__device__ __forceinline__ void wait_async() {
#if __has_builtin(__builtin_amdgcn_s_wait_asynccnt)
    __builtin_amdgcn_s_wait_asynccnt(N);
#else
    asm volatile("s_wait_asynccnt %0" ::"i"(N) : "memory");
#endif
}

// Block tile BM=64, BN=128, BK=64. 256 threads = 8 wave32 waves in a 2x4 grid;
// each wave computes a 32x32 tile = 2x2 WMMA 16x16x32 accumulators.
__global__ __launch_bounds__(256)
void gemm_lif(const __bf16* __restrict__ A, long a_stride,   // [M][K'] bf16
              const __bf16* __restrict__ W, int K, int N,    // [N][K] bf16
              float* __restrict__ v,                         // [M][N] membrane
              float* __restrict__ sout, long s_row_stride, long s_col_off,
              __bf16* __restrict__ sbf)                      // optional [M][N]
{
    __shared__ __attribute__((aligned(16))) __bf16 lA[2][64][PITCH];
    __shared__ __attribute__((aligned(16))) __bf16 lB[2][128][PITCH];

    const int tid  = threadIdx.x;
    const int lane = tid & 31;
    const int wave = tid >> 5;
    const int wm   = wave & 1;     // 0..1 : 32-row group
    const int wn   = wave >> 1;    // 0..3 : 32-col group
    const int lr   = lane & 15;
    const int hs   = lane >> 4;    // lane-half select

    const long bm0 = (long)blockIdx.y * 64;
    const long bn0 = (long)blockIdx.x * 128;

    // 6 x 16B async copies per thread per stage (A: 8KB, B: 16KB)
    auto stage = [&](int buf, int k0) {
        {
            const int r = tid >> 2;            // 0..63
            const int c = (tid & 3) * 16;      // 0,16,32,48
            const __bf16* src = A + (bm0 + r) * a_stride + k0 + c;
            cp16(src,     &lA[buf][r][c]);
            cp16(src + 8, &lA[buf][r][c + 8]);
        }
        {
            const int r = tid >> 1;            // 0..127
            const int c = (tid & 1) * 32;      // 0,32
            const __bf16* src = W + (bn0 + r) * (long)K + k0 + c;
            cp16(src,      &lB[buf][r][c]);
            cp16(src + 8,  &lB[buf][r][c + 8]);
            cp16(src + 16, &lB[buf][r][c + 16]);
            cp16(src + 24, &lB[buf][r][c + 24]);
        }
    };

    v8f acc[2][2] = {};
    const int NK = K / BKT;

    stage(0, 0);
    for (int it = 0; it < NK; ++it) {
        const int cur = it & 1;
        if (it + 1 < NK) {                 // prefetch next tile into other buffer
            stage(cur ^ 1, (it + 1) * BKT);
            wait_async<6>();               // this wave's copies for `cur` landed
        } else {
            wait_async<0>();
        }
        __syncthreads();                   // all waves' copies for `cur` landed

        #pragma unroll
        for (int kk = 0; kk < BKT; kk += 32) {
            // Fragments per documented 16-bit VGPR layouts (ISA 7.12.2):
            // A: lanes 0-15 hold K 0-7/16-23, lanes 16-31 hold K 8-15/24-31.
            // B: lanes 0-15 hold K 0-7/16-23, lanes 16-31 hold K 8-15/24-31.
            v16bf afr[2], bfr[2];
            const int abase = kk + hs * 8;
            #pragma unroll
            for (int i = 0; i < 2; ++i) {
                const int r = wm * 32 + i * 16 + lr;
                v8bf lo = *(const v8bf*)&lA[cur][r][abase];
                v8bf hi = *(const v8bf*)&lA[cur][r][abase + 16];
                #pragma unroll
                for (int j = 0; j < 8; ++j) { afr[i][j] = lo[j]; afr[i][8 + j] = hi[j]; }
            }
            #pragma unroll
            for (int i = 0; i < 2; ++i) {
                const int c = wn * 32 + i * 16 + lr;
                v8bf lo = *(const v8bf*)&lB[cur][c][kk + hs * 8];
                v8bf hi = *(const v8bf*)&lB[cur][c][kk + hs * 8 + 16];
                #pragma unroll
                for (int j = 0; j < 8; ++j) { bfr[i][j] = lo[j]; bfr[i][8 + j] = hi[j]; }
            }
            #pragma unroll
            for (int mi = 0; mi < 2; ++mi)
                #pragma unroll
                for (int ni = 0; ni < 2; ++ni)
                    acc[mi][ni] = __builtin_amdgcn_wmma_f32_16x16x32_bf16(
                        false, afr[mi], false, bfr[ni], (short)0, acc[mi][ni],
                        false, false);
        }
        __syncthreads();                   // done reading `cur`; it may be rewritten
    }

    // Fused LIF: v += (o - v)/2 ; fire = (v >= 0.5) ; hard reset; emit spike.
    #pragma unroll
    for (int mi = 0; mi < 2; ++mi) {
        #pragma unroll
        for (int ni = 0; ni < 2; ++ni) {
            const long n = bn0 + wn * 32 + ni * 16 + lr;
            #pragma unroll
            for (int i = 0; i < 8; ++i) {
                const long m = bm0 + wm * 32 + mi * 16 + hs * 8 + i;
                const float o = acc[mi][ni][i];
                const long vidx = m * (long)N + n;
                float vv = v[vidx];
                vv = vv + (o - vv) * 0.5f;
                const bool fire = (vv >= 0.5f);
                v[vidx] = fire ? 0.0f : vv;
                sout[m * s_row_stride + s_col_off + n] = fire ? 1.0f : 0.0f;
                if (sbf) sbf[vidx] = (__bf16)(fire ? 1.0f : 0.0f);
            }
        }
    }
}

__global__ __launch_bounds__(256) void cvt_f32_bf16(const float* __restrict__ in,
                                                    __bf16* __restrict__ out, int n4) {
    int i = blockIdx.x * blockDim.x + threadIdx.x;
    if (i < n4) {
        v4f f = ((const v4f*)in)[i];
        v4bf h;
        #pragma unroll
        for (int j = 0; j < 4; ++j) h[j] = (__bf16)f[j];
        ((v4bf*)out)[i] = h;
    }
}

// Convert time-slice x[:, t, :] (f32, row stride T*FIN) -> contiguous bf16 [B][FIN].
__global__ __launch_bounds__(256) void cvt_slice(const float* __restrict__ x,
                                                 __bf16* __restrict__ xb, int t) {
    const long i = (long)blockIdx.x * blockDim.x + threadIdx.x;   // over B*FIN/4
    const long b = i / (FIN / 4);
    const long c = i % (FIN / 4);
    v4f f = *(const v4f*)(x + b * (long)T_DIM * FIN + (long)t * FIN + c * 4);
    v4bf h;
    #pragma unroll
    for (int j = 0; j < 4; ++j) h[j] = (__bf16)f[j];
    ((v4bf*)xb)[i] = h;
}

__global__ __launch_bounds__(256) void zero_f32(float* __restrict__ p, int n4) {
    int i = blockIdx.x * blockDim.x + threadIdx.x;
    if (i < n4) ((v4f*)p)[i] = v4f{0.f, 0.f, 0.f, 0.f};
}

extern "C" void kernel_launch(void* const* d_in, const int* in_sizes, int n_in,
                              void* d_out, int out_size, void* d_ws, size_t ws_size,
                              hipStream_t stream) {
    (void)in_sizes; (void)n_in; (void)out_size; (void)ws_size;

    const float* x  = (const float*)d_in[0];   // [B,T,FIN]
    const float* W1 = (const float*)d_in[1];   // [HID,FIN]
    const float* W2 = (const float*)d_in[2];   // [OUTN,HID]

    float* out = (float*)d_out;
    float* com = out;                                     // [B,T,OUTN]
    float* ch  = com + (size_t)B_DIM * T_DIM * OUTN;      // [B,T,HID]
    float* xo  = ch  + (size_t)B_DIM * T_DIM * HID;       // [B,T,FIN] passthrough

    // workspace layout (22 MB total)
    char* ws = (char*)d_ws;
    const size_t OFF_W2 = (size_t)HID * FIN * 2;              //  8 MB
    const size_t OFF_V1 = OFF_W2 + (size_t)OUTN * HID * 2;    // 12 MB
    const size_t OFF_V2 = OFF_V1 + (size_t)B_DIM * HID * 4;   // 16 MB
    const size_t OFF_S1 = OFF_V2 + (size_t)B_DIM * OUTN * 4;  // 18 MB
    const size_t OFF_XB = OFF_S1 + (size_t)B_DIM * HID * 2;   // 20 MB
    __bf16* w1b = (__bf16*)ws;
    __bf16* w2b = (__bf16*)(ws + OFF_W2);
    float*  v1  = (float*)(ws + OFF_V1);
    float*  v2  = (float*)(ws + OFF_V2);
    __bf16* s1b = (__bf16*)(ws + OFF_S1);
    __bf16* xb  = (__bf16*)(ws + OFF_XB);

    // weights -> bf16 once per launch (L2-resident across the whole T loop)
    {
        int n4 = HID * FIN / 4;
        cvt_f32_bf16<<<n4 / 256, 256, 0, stream>>>(W1, w1b, n4);
        n4 = OUTN * HID / 4;
        cvt_f32_bf16<<<n4 / 256, 256, 0, stream>>>(W2, w2b, n4);
    }
    // v1,v2 are contiguous: zero both in one pass
    {
        int n4 = (B_DIM * HID + B_DIM * OUTN) / 4;
        zero_f32<<<(n4 + 255) / 256, 256, 0, stream>>>(v1, n4);
    }
    // x passthrough output
    (void)hipMemcpyAsync(xo, x, (size_t)B_DIM * T_DIM * FIN * sizeof(float),
                         hipMemcpyDeviceToDevice, stream);

    const int nslice4 = B_DIM * FIN / 4;
    for (int t = 0; t < T_DIM; ++t) {
        // x_t -> bf16 (binary spikes: exact)
        cvt_slice<<<nslice4 / 256, 256, 0, stream>>>(x, xb, t);
        // layer 1: h = x_t @ W1^T ; LIF -> s1 (f32 into ch, bf16 into ws)
        gemm_lif<<<dim3(HID / 128, B_DIM / 64), 256, 0, stream>>>(
            xb, (long)FIN, w1b, FIN, HID, v1,
            ch, (long)T_DIM * HID, (long)t * HID, s1b);
        // layer 2: o = s1 @ W2^T ; LIF -> s2 (f32 into com)
        gemm_lif<<<dim3(OUTN / 128, B_DIM / 64), 256, 0, stream>>>(
            s1b, (long)HID, w2b, HID, OUTN, v2,
            com, (long)T_DIM * OUTN, (long)t * OUTN, nullptr);
    }
}
